// ProperTransformer_59476707115071
// MI455X (gfx1250) — matrix operand order repaired and checked by reading, not compiled, not run on hardware
//
#include <hip/hip_runtime.h>
#include <hip/hip_bf16.h>

#define VV   32000
#define DD   512
#define HH   8
#define LLAY 4
#define DFFC 2048
#define BBAT 2
#define SSEQ 2048
#define DKC  64
#define MM   (BBAT * SSEQ)   // 4096 tokens

typedef __attribute__((ext_vector_type(16))) _Float16 v16h;
typedef __attribute__((ext_vector_type(8)))  _Float16 v8h;
typedef __attribute__((ext_vector_type(8)))  float    v8f;
typedef __attribute__((ext_vector_type(4)))  unsigned int u32x4;
typedef __attribute__((ext_vector_type(8)))  unsigned int u32x8;

// ---------------------------------------------------------------------------
// CDNA5 async global->LDS copy (ASYNCcnt-tracked, CDNA5 ISA)
// ---------------------------------------------------------------------------
__device__ __forceinline__ void async_copy_b128(unsigned int lds, const void* g) {
  asm volatile("global_load_async_to_lds_b128 %0, %1, off"
               :: "v"(lds), "v"((unsigned long long)(uintptr_t)g)
               : "memory");
}
template <int N>
__device__ __forceinline__ void wait_asynccnt() {
  asm volatile("s_wait_asynccnt %0" :: "i"(N) : "memory");
}

// ---------------------------------------------------------------------------
// CDNA5 Tensor Data Mover: DMA a 2D tile (64 rows x 32 halfs, row stride Kdim)
// from global into LDS. D# built per CDNA5 ISA;
// 2-SGPR-group form (tensors up to 2D). TENSORcnt-tracked.
// ---------------------------------------------------------------------------
__device__ __forceinline__ void tdm_load_b_tile(unsigned int lds_addr,
                                                const void* gaddr,
                                                int Kdim, int Ndim) {
  unsigned long long ga = (unsigned long long)(uintptr_t)gaddr;
  u32x4 g0;
  g0[0] = 1u;                                        // count=1, user descriptor
  g0[1] = lds_addr;                                  // lds_addr (bytes)
  g0[2] = (unsigned int)(ga & 0xffffffffu);          // global_addr[31:0]
  g0[3] = (unsigned int)((ga >> 32) & 0x01ffffffu)   // global_addr[56:32]
        | (2u << 30);                                // type = 2 ("image")
  u32x8 g1;
  g1[0] = 1u << 16;                                  // data_size=1 -> 2 bytes
  g1[1] = ((unsigned int)Kdim & 0xffffu) << 16;      // tensor_dim0[15:0]
  g1[2] = (((unsigned int)Kdim >> 16) & 0xffffu)     // tensor_dim0[31:16]
        | (((unsigned int)Ndim & 0xffffu) << 16);    // tensor_dim1[15:0]
  g1[3] = (((unsigned int)Ndim >> 16) & 0xffffu)     // tensor_dim1[31:16]
        | (32u << 16);                               // tile_dim0 = 32 elems
  g1[4] = 64u;                                       // tile_dim1 = 64 rows
  g1[5] = (unsigned int)Kdim;                        // tensor_dim0_stride[31:0]
  g1[6] = 0u;
  g1[7] = 0u;
  asm volatile("tensor_load_to_lds %0, %1" :: "s"(g0), "s"(g1) : "memory");
}

// ---------------------------------------------------------------------------
// Fragment loaders: A and B share contiguous k order (not the ISA table order)
// A (16x32 f16): lane L (hl = L/16, m = L%16) holds
//   elem i -> A[m, kbase + hl*16 + i]   (contiguous 32 bytes)
// B (32x16 f16) from a row-major [N,K] ("Wt") matrix: lane L (n = L%16, kh=L/16)
//   elem i -> Wt[n, kbase + kh*16 + i]   (contiguous 32 bytes)
// C/D (16x16 f32): reg r, lanes 0-15: (M=r, N=lane); lanes 16-31: (M=r+8, N=lane-16)
// ---------------------------------------------------------------------------
__device__ __forceinline__ v16h load_a_frag(const _Float16* A, int lda, int m,
                                            int kbase, int hl) {
  const _Float16* p = A + (size_t)m * lda + kbase + hl * 16;
  v8h lo = *(const v8h*)p;
  v8h hi = *(const v8h*)(p + 8);
  v16h a;
#pragma unroll
  for (int i = 0; i < 8; ++i) { a[i] = lo[i]; a[i + 8] = hi[i]; }
  return a;
}

__device__ __forceinline__ v16h load_b_frag(const _Float16* Wt, int K, int ncol,
                                            int kbase, int kh) {
  return *(const v16h*)(Wt + (size_t)ncol * K + kbase + kh * 16);
}

__device__ __forceinline__ v8f wmma16(v16h a, v16h b, v8f c) {
  return __builtin_amdgcn_wmma_f32_16x16x32_f16(false, a, false, b, (short)0, c,
                                                false, false);
}

__device__ __forceinline__ float gelu_exact(float x) {
  return 0.5f * x * (1.0f + erff(x * 0.70710678118654752f));
}

// ---------------------------------------------------------------------------
// Weight conversion: f32 [K,N] row-major -> f16 [N,K] row-major (transposed)
// ---------------------------------------------------------------------------
__global__ void transpose_f32_to_f16(const float* __restrict__ in,
                                     _Float16* __restrict__ out, int K, int N) {
  int idx = blockIdx.x * 256 + threadIdx.x;
  if (idx >= K * N) return;
  int k = idx / N, n = idx % N;
  out[(size_t)n * K + k] = (_Float16)in[idx];
}

// ---------------------------------------------------------------------------
// Embedding: x = tok_emb[tok] * sqrt(D) + pe[s]
// ---------------------------------------------------------------------------
__global__ void embed_kernel(const int* __restrict__ tokens,
                             const float* __restrict__ tok_emb,
                             const float* __restrict__ pe,
                             float* __restrict__ x) {
  int i = blockIdx.x * 256 + threadIdx.x;
  if (i >= MM * DD) return;
  int row = i / DD, d = i % DD;
  int s = row % SSEQ;
  int tok = tokens[row];
  x[i] = tok_emb[(size_t)tok * DD + d] * 22.627416997969522f + pe[(size_t)s * DD + d];
}

// ---------------------------------------------------------------------------
// LayerNorm (one 128-thread block per token row, D=512) -> f16 output
// ---------------------------------------------------------------------------
__global__ __launch_bounds__(128) void layernorm_kernel(
    const float* __restrict__ x, const float* __restrict__ sc,
    const float* __restrict__ bi, _Float16* __restrict__ out) {
  __shared__ float rs[128], rq[128];
  int row = blockIdx.x, tid = threadIdx.x;
  const float* xr = x + (size_t)row * DD;
  float v[4], s = 0.f, q = 0.f;
#pragma unroll
  for (int i = 0; i < 4; ++i) {
    float t = xr[tid + i * 128];
    v[i] = t; s += t; q += t * t;
  }
  rs[tid] = s; rq[tid] = q; __syncthreads();
  for (int off = 64; off > 0; off >>= 1) {
    if (tid < off) { rs[tid] += rs[tid + off]; rq[tid] += rq[tid + off]; }
    __syncthreads();
  }
  float mu  = rs[0] * (1.0f / DD);
  float var = rq[0] * (1.0f / DD) - mu * mu;
  float r   = rsqrtf(var + 1e-5f);
  _Float16* orow = out + (size_t)row * DD;
#pragma unroll
  for (int i = 0; i < 4; ++i) {
    int d = tid + i * 128;
    orow[d] = (_Float16)((v[i] - mu) * r * sc[d] + bi[d]);
  }
}

// ---------------------------------------------------------------------------
// WMMA GEMM, double-buffered LDS staging using BOTH CDNA5 DMA paths:
//   A tile (128x32 f16): GLOBAL_LOAD_ASYNC_TO_LDS_B128 (2 per lane, ASYNCcnt)
//   B tile (64x32 f16):  TENSOR_LOAD_TO_LDS (wave 0, TENSORcnt)
// Block tile 128(M) x 64(N); 8 waves in 4(M) x 2(N); wave tile 32x32.
// ---------------------------------------------------------------------------
enum { EPI_F32OUT = 0, EPI_RESID = 1, EPI_GELU = 2, EPI_QKV = 3 };

__device__ __forceinline__ void qkv_scatter(_Float16* q, _Float16* k,
                                            _Float16* vt, int row, int col,
                                            float v) {
  int b = row / SSEQ, s = row % SSEQ;
  int which = col >> 9;        // 0:q 1:k 2:v
  int d = col & 511;
  int h = d >> 6, dk = d & 63;
  _Float16 hv = (_Float16)v;
  size_t bh = (size_t)b * HH + h;
  if (which == 0)      q [(bh * SSEQ + s) * DKC + dk] = hv;
  else if (which == 1) k [(bh * SSEQ + s) * DKC + dk] = hv;
  else                 vt[(bh * DKC  + dk) * SSEQ + s] = hv;   // V^T layout
}

template <int EPI>
__global__ __launch_bounds__(256) void gemm_wmma(
    const _Float16* __restrict__ A, const _Float16* __restrict__ Wt,
    const float* __restrict__ bias, int Mdim, int Ndim, int Kdim,
    float* __restrict__ out_f32, const float* __restrict__ resid,
    _Float16* __restrict__ out_f16, _Float16* __restrict__ qb,
    _Float16* __restrict__ kb, _Float16* __restrict__ vtb) {
  (void)Mdim;
  __shared__ __align__(32) _Float16 Asm[2][128 * 32];
  __shared__ __align__(32) _Float16 Bsm[2][64 * 32];

  const int tid  = threadIdx.x;
  const int lane = tid & 31;
  const int wave = tid >> 5;
  const int hl = lane >> 4;     // half-wave id
  const int ln = lane & 15;
  const int tileM = blockIdx.y * 128;
  const int tileN = blockIdx.x * 64;
  const int waveM = (wave & 3) * 32;
  const int waveN = (wave >> 2) * 32;

  // Stage t: A via per-lane async copies, B via one TDM descriptor (wave 0).
  auto issue_stage = [&](int t) {
    const int kb_ = t * 32;
    const int buf = t & 1;
#pragma unroll
    for (int i = 0; i < 2; ++i) {           // A tile: 512 16B chunks
      int c = tid + 256 * i;
      int row = c >> 2, co = (c & 3) * 8;
      async_copy_b128((unsigned int)(uintptr_t)&Asm[buf][row * 32 + co],
                      A + (size_t)(tileM + row) * Kdim + kb_ + co);
    }
    if (wave == 0) {                         // B tile: one TDM 2D tile load
      tdm_load_b_tile((unsigned int)(uintptr_t)&Bsm[buf][0],
                      Wt + (size_t)tileN * Kdim + kb_, Kdim, Ndim);
    }
  };

  v8f cacc[2][2] = {};
  const int nsteps = Kdim / 32;
  issue_stage(0);
  for (int t = 0; t < nsteps; ++t) {
    if (t + 1 < nsteps) {
      issue_stage(t + 1);          // prefetch next stage into other buffer
      wait_asynccnt<2>();          // oldest 2 (stage t A) complete; in-order
      if (wave == 0) __builtin_amdgcn_s_wait_tensorcnt(1);
    } else {
      wait_asynccnt<0>();
      if (wave == 0) __builtin_amdgcn_s_wait_tensorcnt(0);
    }
    __syncthreads();               // all waves' stage-t data visible
    const _Float16* Ab = &Asm[t & 1][0];
    const _Float16* Bb = &Bsm[t & 1][0];
    v16h a0 = load_a_frag(Ab, 32, waveM + ln,      0, hl);
    v16h a1 = load_a_frag(Ab, 32, waveM + 16 + ln, 0, hl);
    v16h b0 = load_b_frag(Bb, 32, waveN + ln,      0, hl);
    v16h b1 = load_b_frag(Bb, 32, waveN + 16 + ln, 0, hl);
    cacc[0][0] = wmma16(a0, b0, cacc[0][0]);
    cacc[0][1] = wmma16(a0, b1, cacc[0][1]);
    cacc[1][0] = wmma16(a1, b0, cacc[1][0]);
    cacc[1][1] = wmma16(a1, b1, cacc[1][1]);
    __syncthreads();               // done reading buf before it is re-staged
  }

#pragma unroll
  for (int mi = 0; mi < 2; ++mi) {
#pragma unroll
    for (int ni = 0; ni < 2; ++ni) {
#pragma unroll
      for (int r = 0; r < 8; ++r) {
        int mrow = tileM + waveM + mi * 16 + r + hl * 8;
        int col  = tileN + waveN + ni * 16 + ln;
        float v = cacc[mi][ni][r] + bias[col];
        size_t idx = (size_t)mrow * Ndim + col;
        if (EPI == EPI_F32OUT) {
          out_f32[idx] = v;
        } else if (EPI == EPI_RESID) {
          out_f32[idx] = resid[idx] + v;
        } else if (EPI == EPI_GELU) {
          out_f16[idx] = (_Float16)gelu_exact(v);
        } else {
          qkv_scatter(qb, kb, vtb, mrow, col, v);
        }
      }
    }
  }
}

// ---------------------------------------------------------------------------
// Flash-style causal attention: one wave per (b, h, 16-row q tile).
// QK^T and P@V via WMMA; online softmax with half-wave shuffle reductions.
// ---------------------------------------------------------------------------
__global__ __launch_bounds__(32) void attention_kernel(
    const _Float16* __restrict__ q, const _Float16* __restrict__ kmat,
    const _Float16* __restrict__ vt, _Float16* __restrict__ out) {
  __shared__ __align__(64) _Float16 pls[16 * 32];   // P tile staging (C->A relayout)
  const int lane = threadIdx.x & 31;
  const int hl = lane >> 4, ln = lane & 15;
  const int qt = blockIdx.x, h = blockIdx.y, b = blockIdx.z;
  const int qbase = qt * 16;
  const float SCALE = 0.125f;   // 1/sqrt(64)

  const _Float16* qp = q    + (((size_t)b * HH + h) * SSEQ) * DKC;
  const _Float16* kp = kmat + (((size_t)b * HH + h) * SSEQ) * DKC;
  const _Float16* vp = vt   + (((size_t)b * HH + h) * DKC) * SSEQ;

  // Q fragments for K-dim 0..31 and 32..63 (loaded once, reused)
  v16h aq0 = load_a_frag(qp, DKC, qbase + ln, 0, hl);
  v16h aq1 = load_a_frag(qp, DKC, qbase + ln, 32, hl);

  v8f o0 = {}, o1 = {}, o2 = {}, o3 = {};
  float mrow[8], lrow[8];
#pragma unroll
  for (int r = 0; r < 8; ++r) { mrow[r] = -1e30f; lrow[r] = 0.f; }

  const int nkb = (qbase + 16 + 31) >> 5;   // 32-key blocks (causal bound)
  for (int j = 0; j < nkb; ++j) {
    const int kcol = j * 32;
    if (j + 1 < nkb) {
      __builtin_prefetch(kp + (size_t)(kcol + 32 + ln) * DKC, 0, 1);
      __builtin_prefetch(vp + (size_t)ln * SSEQ + kcol + 32, 0, 1);
    }
    // Issue all four K-tile fragments into distinct registers, then WMMA,
    // letting the compiler clause the loads and use partial loadcnt waits.
    v16h bk00 = load_b_frag(kp, DKC, kcol + ln,      0,  hl);
    v16h bk10 = load_b_frag(kp, DKC, kcol + 16 + ln, 0,  hl);
    v16h bk01 = load_b_frag(kp, DKC, kcol + ln,      32, hl);
    v16h bk11 = load_b_frag(kp, DKC, kcol + 16 + ln, 32, hl);
    v8f s0 = {}, s1 = {};
    s0 = wmma16(aq0, bk00, s0);
    s1 = wmma16(aq0, bk10, s1);
    s0 = wmma16(aq1, bk01, s0);
    s1 = wmma16(aq1, bk11, s1);

    float alpha[8];
#pragma unroll
    for (int r = 0; r < 8; ++r) {
      int row = qbase + r + hl * 8;
      float x0 = (kcol + ln <= row)      ? s0[r] * SCALE : -1e30f;
      float x1 = (kcol + 16 + ln <= row) ? s1[r] * SCALE : -1e30f;
      float t = fmaxf(x0, x1);
      for (int off = 1; off < 16; off <<= 1) t = fmaxf(t, __shfl_xor(t, off, 32));
      float nm = fmaxf(mrow[r], t);
      float a_ = expf(mrow[r] - nm);
      mrow[r] = nm;
      float p0 = expf(x0 - nm), p1 = expf(x1 - nm);
      float ps = p0 + p1;
      for (int off = 1; off < 16; off <<= 1) ps += __shfl_xor(ps, off, 32);
      lrow[r] = lrow[r] * a_ + ps;
      alpha[r] = a_;
      pls[(r + hl * 8) * 32 + ln]      = (_Float16)p0;
      pls[(r + hl * 8) * 32 + 16 + ln] = (_Float16)p1;
    }
#pragma unroll
    for (int r = 0; r < 8; ++r) {
      o0[r] *= alpha[r]; o1[r] *= alpha[r];
      o2[r] *= alpha[r]; o3[r] *= alpha[r];
    }
    __syncthreads();
    v16h ap = load_a_frag(pls, 32, ln, 0, hl);   // P tile in A layout
    v16h bv0 = load_b_frag(vp, SSEQ, 0  + ln, kcol, hl);
    v16h bv1 = load_b_frag(vp, SSEQ, 16 + ln, kcol, hl);
    v16h bv2 = load_b_frag(vp, SSEQ, 32 + ln, kcol, hl);
    v16h bv3 = load_b_frag(vp, SSEQ, 48 + ln, kcol, hl);
    o0 = wmma16(ap, bv0, o0);
    o1 = wmma16(ap, bv1, o1);
    o2 = wmma16(ap, bv2, o2);
    o3 = wmma16(ap, bv3, o3);
    __syncthreads();
  }

#pragma unroll
  for (int r = 0; r < 8; ++r) {
    float inv = 1.0f / lrow[r];
    int row = qbase + r + hl * 8;
    size_t base = ((size_t)b * SSEQ + row) * DD + (size_t)h * DKC;
    out[base + 0  + ln] = (_Float16)(o0[r] * inv);
    out[base + 16 + ln] = (_Float16)(o1[r] * inv);
    out[base + 32 + ln] = (_Float16)(o2[r] * inv);
    out[base + 48 + ln] = (_Float16)(o3[r] * inv);
  }
}

// ---------------------------------------------------------------------------
// Host orchestration
// ---------------------------------------------------------------------------
extern "C" void kernel_launch(void* const* d_in, const int* in_sizes, int n_in,
                              void* d_out, int out_size, void* d_ws,
                              size_t ws_size, hipStream_t stream) {
  (void)in_sizes; (void)n_in; (void)out_size; (void)ws_size;
  const int*   tokens  = (const int*)d_in[0];
  const float* tok_emb = (const float*)d_in[1];
  const float* pe      = (const float*)d_in[2];
  const float* ln1_s   = (const float*)d_in[3];
  const float* ln1_b   = (const float*)d_in[4];
  const float* w_qkv   = (const float*)d_in[5];
  const float* b_qkv   = (const float*)d_in[6];
  const float* w_o     = (const float*)d_in[7];
  const float* b_o     = (const float*)d_in[8];
  const float* ln2_s   = (const float*)d_in[9];
  const float* ln2_b   = (const float*)d_in[10];
  const float* w1      = (const float*)d_in[11];
  const float* b1      = (const float*)d_in[12];
  const float* w2      = (const float*)d_in[13];
  const float* b2      = (const float*)d_in[14];
  const float* lnf_s   = (const float*)d_in[15];
  const float* lnf_b   = (const float*)d_in[16];
  const float* head_w  = (const float*)d_in[17];
  const float* head_b  = (const float*)d_in[18];

  char* ws = (char*)d_ws;
  size_t off = 0;
  auto alloc = [&](size_t bytes) -> char* {
    char* p = ws + off;
    off = (off + bytes + 255) & ~(size_t)255;
    return p;
  };
  _Float16* wqkv_t = (_Float16*)alloc((size_t)LLAY * 3 * DD * DD * 2);
  _Float16* wo_t   = (_Float16*)alloc((size_t)LLAY * DD * DD * 2);
  _Float16* w1_t   = (_Float16*)alloc((size_t)LLAY * DFFC * DD * 2);
  _Float16* w2_t   = (_Float16*)alloc((size_t)LLAY * DD * DFFC * 2);
  _Float16* head_t = (_Float16*)alloc((size_t)VV * DD * 2);
  float*    x      = (float*)   alloc((size_t)MM * DD * 4);
  _Float16* hbuf   = (_Float16*)alloc((size_t)MM * DD * 2);
  _Float16* qbuf   = (_Float16*)alloc((size_t)BBAT * HH * SSEQ * DKC * 2);
  _Float16* kbuf   = (_Float16*)alloc((size_t)BBAT * HH * SSEQ * DKC * 2);
  _Float16* vtbuf  = (_Float16*)alloc((size_t)BBAT * HH * DKC * SSEQ * 2);
  _Float16* atbuf  = (_Float16*)alloc((size_t)MM * DD * 2);
  _Float16* ffbuf  = (_Float16*)alloc((size_t)MM * DFFC * 2);

  // --- weight conversion (f32 [K,N] -> f16 [N,K]) ---
  for (int l = 0; l < LLAY; ++l) {
    transpose_f32_to_f16<<<(DD * 3 * DD + 255) / 256, 256, 0, stream>>>(
        w_qkv + (size_t)l * DD * 3 * DD, wqkv_t + (size_t)l * 3 * DD * DD, DD, 3 * DD);
    transpose_f32_to_f16<<<(DD * DD + 255) / 256, 256, 0, stream>>>(
        w_o + (size_t)l * DD * DD, wo_t + (size_t)l * DD * DD, DD, DD);
    transpose_f32_to_f16<<<(DD * DFFC + 255) / 256, 256, 0, stream>>>(
        w1 + (size_t)l * DD * DFFC, w1_t + (size_t)l * DFFC * DD, DD, DFFC);
    transpose_f32_to_f16<<<(DFFC * DD + 255) / 256, 256, 0, stream>>>(
        w2 + (size_t)l * DFFC * DD, w2_t + (size_t)l * DD * DFFC, DFFC, DD);
  }
  transpose_f32_to_f16<<<(DD * VV + 255) / 256, 256, 0, stream>>>(
      head_w, head_t, DD, VV);

  // --- embedding ---
  embed_kernel<<<(MM * DD + 255) / 256, 256, 0, stream>>>(tokens, tok_emb, pe, x);

  // --- transformer layers ---
  for (int l = 0; l < LLAY; ++l) {
    layernorm_kernel<<<MM, 128, 0, stream>>>(x, ln1_s + l * DD, ln1_b + l * DD, hbuf);
    gemm_wmma<EPI_QKV><<<dim3(3 * DD / 64, MM / 128), 256, 0, stream>>>(
        hbuf, wqkv_t + (size_t)l * 3 * DD * DD, b_qkv + (size_t)l * 3 * DD,
        MM, 3 * DD, DD, nullptr, nullptr, nullptr, qbuf, kbuf, vtbuf);
    attention_kernel<<<dim3(SSEQ / 16, HH, BBAT), 32, 0, stream>>>(
        qbuf, kbuf, vtbuf, atbuf);
    gemm_wmma<EPI_RESID><<<dim3(DD / 64, MM / 128), 256, 0, stream>>>(
        atbuf, wo_t + (size_t)l * DD * DD, b_o + (size_t)l * DD,
        MM, DD, DD, x, x, nullptr, nullptr, nullptr, nullptr);
    layernorm_kernel<<<MM, 128, 0, stream>>>(x, ln2_s + l * DD, ln2_b + l * DD, hbuf);
    gemm_wmma<EPI_GELU><<<dim3(DFFC / 64, MM / 128), 256, 0, stream>>>(
        hbuf, w1_t + (size_t)l * DFFC * DD, b1 + (size_t)l * DFFC,
        MM, DFFC, DD, nullptr, nullptr, ffbuf, nullptr, nullptr, nullptr);
    gemm_wmma<EPI_RESID><<<dim3(DD / 64, MM / 128), 256, 0, stream>>>(
        ffbuf, w2_t + (size_t)l * DD * DFFC, b2 + (size_t)l * DD,
        MM, DD, DFFC, x, x, nullptr, nullptr, nullptr, nullptr);
  }

  // --- final LN + vocab head (f32 logits to d_out) ---
  layernorm_kernel<<<MM, 128, 0, stream>>>(x, lnf_s, lnf_b, hbuf);
  gemm_wmma<EPI_F32OUT><<<dim3(VV / 64, MM / 128), 256, 0, stream>>>(
      hbuf, head_t, head_b, MM, VV, DD, (float*)d_out, nullptr, nullptr,
      nullptr, nullptr, nullptr);
}
